// QuantLinear_63058709840002
// MI455X (gfx1250) — matrix operand order repaired and checked by reading, not compiled, not run on hardware
//
#include <hip/hip_runtime.h>

// ---------------------------------------------------------------------------
// QuIP#-style quantized linear for MI455X (gfx1250, wave32, WMMA).
//   xh = FWHT(x*SU) * (1/sqrt(8192) * Wscale)
//   y  = xh @ decode(Qidxs, grid).T        (64 x 8192 x 8192 GEMM, bf16 WMMA,
//                                           async global->LDS double buffering)
//   out= FWHT(y) * (1/sqrt(8192)) * SV + bias
// ---------------------------------------------------------------------------

#define TOKENS 64
#define IN_F   8192
#define OUT_F  8192
#define CODESZ 8
#define CB_SIZE 256

typedef __attribute__((ext_vector_type(16))) __bf16 v16bf;
typedef __attribute__((ext_vector_type(8)))  __bf16 v8bf;
typedef __attribute__((ext_vector_type(8)))  float  v8f;

#define INV_SQRT_8192 0.011048543456039806f

// ------------------------- CDNA5 async helpers -----------------------------
// GLOBAL_LOAD_ASYNC_TO_LDS_B128: per-lane 16B DMA global->LDS, ASYNCcnt.
static __device__ inline void async_copy16(unsigned lds_off, const void* gptr)
{
    asm volatile("global_load_async_to_lds_b128 %0, %1, off"
                 :: "v"(lds_off), "v"((unsigned long long)(uintptr_t)gptr)
                 : "memory");
}
static __device__ inline void wait_async0()
{
    asm volatile("s_wait_asynccnt 0x0" ::: "memory");
}

// ------------------------- FWHT input kernel -------------------------------
__global__ __launch_bounds__(256)
void fwht_in_kernel(const float* __restrict__ x, const float* __restrict__ SU,
                    const float* __restrict__ Wscale, __bf16* __restrict__ xh)
{
    __shared__ float buf[IN_F];
    const int row = blockIdx.x;
    const int t   = threadIdx.x;

    for (int i = t; i < IN_F; i += 256)
        buf[i] = x[row * IN_F + i] * SU[i];
    __syncthreads();

    for (int h = 1; h < IN_F; h <<= 1) {
        for (int p = t; p < IN_F / 2; p += 256) {
            int i = ((p & ~(h - 1)) << 1) + (p & (h - 1));
            int j = i + h;
            float a = buf[i], b = buf[j];
            buf[i] = a + b;
            buf[j] = a - b;
        }
        __syncthreads();
    }

    const float s = INV_SQRT_8192 * Wscale[0];
    for (int i = t; i < IN_F; i += 256)
        xh[row * IN_F + i] = (__bf16)(buf[i] * s);
}

// ------------------------- GEMM with codebook decode -----------------------
#define KC  128           // K chunk per LDS stage
#define LDT (KC + 8)      // padded LDS row: 136 bf16 = 272 B (16B aligned)

static __device__ inline v16bf combine8(v8bf lo, v8bf hi)
{
    return __builtin_shufflevector(lo, hi, 0, 1, 2, 3, 4, 5, 6, 7,
                                           8, 9, 10, 11, 12, 13, 14, 15);
}

__global__ __launch_bounds__(256)
void gemm_decode_kernel(const __bf16* __restrict__ xh,    // [64][8192] bf16
                        const int*    __restrict__ Qidxs, // [8192][1024]
                        const float*  __restrict__ grid,  // [256][8]
                        float*        __restrict__ y)     // [64][8192]
{
    __shared__ __bf16 Atile[2][TOKENS * LDT];  // 2 x 17.4 KB
    __shared__ __bf16 Wtile[2][32 * LDT];      // 2 x  8.7 KB
    __shared__ __bf16 gridb[CB_SIZE * CODESZ]; //      4.0 KB   (total ~56 KB)

    const int t      = threadIdx.x;          // 0..255
    const int lane   = t & 31;
    const int wave   = t >> 5;               // 0..7
    const int mtile  = wave & 3;             // M block of 16
    const int ntile  = wave >> 2;            // N block of 16
    const int blockN = blockIdx.x * 32;      // 256 blocks cover N=8192

    // Codebook -> LDS as bf16 (one shot)
    {
        const float* g = grid  + t * 8;
        __bf16*      d = gridb + t * 8;
#pragma unroll
        for (int i = 0; i < 8; ++i) d[i] = (__bf16)g[i];
    }
    __syncthreads();

    v8f acc = {};

    const int m     = lane & 15;   // A: lane -> M row
    const int khalf = lane >> 4;   // A: K half-block select
    const int n     = lane & 15;   // B: lane -> N col
    const int kgrp  = lane >> 4;   // B: K 8-block select in each 16-half

    // ---- stage chunk k0 into parity buffer p (async A copy + W decode) ----
    auto stage = [&](int k0, int p) {
        // A tile: 64 x 128 bf16; 4 threads/row, 32 elems (4 x b128) each.
        {
            const int r = t >> 2;
            const int c = (t & 3) * 32;
            const __bf16* src = xh + r * IN_F + k0 + c;
            unsigned dst = (unsigned)(uintptr_t)&Atile[p][r * LDT + c];
#pragma unroll
            for (int i = 0; i < 4; ++i)
                async_copy16(dst + i * 16, src + i * 8);
        }
        // W tile: 32 rows x 16 codes -> 32 x 128 bf16; 2 codes/thread.
        {
            const int cbase = k0 >> 3;
#pragma unroll
            for (int i = 0; i < 2; ++i) {
                const int cidx = t * 2 + i;      // 0..511
                const int row  = cidx >> 4;      // 0..31
                const int kc   = cidx & 15;      // 0..15
                const int q    = __builtin_nontemporal_load(
                    Qidxs + (blockN + row) * (IN_F / CODESZ) + cbase + kc);
                *(v8bf*)(&Wtile[p][row * LDT + kc * CODESZ]) =
                    *(const v8bf*)(gridb + q * CODESZ);
            }
        }
    };

    // ---- compute on parity buffer p: 4 x WMMA bf16 (K=32 each) per wave ---
    auto compute = [&](int p) {
        const __bf16* arow = &Atile[p][(mtile * 16 + m) * LDT];
        const __bf16* brow = &Wtile[p][(ntile * 16 + n) * LDT];
#pragma unroll
        for (int ks = 0; ks < KC / 32; ++ks) {
            const int kb = ks * 32;
            v16bf a = combine8(*(const v8bf*)(arow + kb + khalf * 8),
                               *(const v8bf*)(arow + kb + 16 + khalf * 8));
            v16bf b = combine8(*(const v8bf*)(brow + kb + kgrp * 8),
                               *(const v8bf*)(brow + kb + 16 + kgrp * 8));
            acc = __builtin_amdgcn_wmma_f32_16x16x32_bf16(
                      false, a, false, b, (short)0, acc, false, false);
        }
    };

    stage(0, 0);   // prologue
    for (int c = 0; c < IN_F / KC; c += 2) {
        wait_async0();
        __syncthreads();
        if (c + 1 < IN_F / KC) stage((c + 1) * KC, 1);  // overlaps compute(0)
        compute(0);
        wait_async0();
        __syncthreads();
        if (c + 2 < IN_F / KC) stage((c + 2) * KC, 0);  // overlaps compute(1)
        compute(1);
    }

    // ---- Store 16x16 f32 tile: VGPR r, lane l -> (M = r + (l>>4)*8, N = l&15)
    const int gm = mtile * 16 + (lane >> 4) * 8;
    const int gn = blockN + ntile * 16 + (lane & 15);
#pragma unroll
    for (int r = 0; r < 8; ++r)
        y[(gm + r) * OUT_F + gn] = acc[r];
}

// ------------------------- FWHT output kernel ------------------------------
__global__ __launch_bounds__(256)
void fwht_out_kernel(const float* __restrict__ y, const float* __restrict__ SV,
                     const float* __restrict__ bias, float* __restrict__ out)
{
    __shared__ float buf[OUT_F];
    const int row = blockIdx.x;
    const int t   = threadIdx.x;

    for (int i = t; i < OUT_F; i += 256)
        buf[i] = y[row * OUT_F + i];
    __syncthreads();

    for (int h = 1; h < OUT_F; h <<= 1) {
        for (int p = t; p < OUT_F / 2; p += 256) {
            int i = ((p & ~(h - 1)) << 1) + (p & (h - 1));
            int j = i + h;
            float a = buf[i], b = buf[j];
            buf[i] = a + b;
            buf[j] = a - b;
        }
        __syncthreads();
    }

    for (int i = t; i < OUT_F; i += 256)
        out[row * OUT_F + i] = buf[i] * INV_SQRT_8192 * SV[i] + bias[i];
}

// ---------------------------------------------------------------------------
extern "C" void kernel_launch(void* const* d_in, const int* in_sizes, int n_in,
                              void* d_out, int out_size, void* d_ws, size_t ws_size,
                              hipStream_t stream)
{
    const float* x      = (const float*)d_in[0];   // (64, 8192)
    const float* SU     = (const float*)d_in[1];   // (8192,)
    const float* SV     = (const float*)d_in[2];   // (8192,)
    const float* grid   = (const float*)d_in[3];   // (256, 8)
    const float* Wscale = (const float*)d_in[4];   // (1,)
    const float* bias   = (const float*)d_in[5];   // (8192,)
    const int*   Qidxs  = (const int*)d_in[6];     // (8192, 1024)
    float*       out    = (float*)d_out;           // (64, 8192) fp32

    // Workspace layout: [0, 1MB) xh bf16; [1MB, 3MB) y fp32
    __bf16* xh = (__bf16*)d_ws;
    float*  y  = (float*)((char*)d_ws + (size_t)TOKENS * IN_F * sizeof(__bf16));

    fwht_in_kernel<<<TOKENS, 256, 0, stream>>>(x, SU, Wscale, xh);
    gemm_decode_kernel<<<OUT_F / 32, 256, 0, stream>>>(xh, Qidxs, grid, y);
    fwht_out_kernel<<<TOKENS, 256, 0, stream>>>(y, SV, bias, out);
}
